// SelfAttention_40106404610690
// MI455X (gfx1250) — matrix operand order repaired and checked by reading, not compiled, not run on hardware
//
#include <hip/hip_runtime.h>
#include <hip/hip_bf16.h>

typedef __attribute__((ext_vector_type(16))) __bf16 v16bf;
typedef __attribute__((ext_vector_type(8)))  __bf16 v8bf;
typedef __attribute__((ext_vector_type(8)))  float  v8f;

#define M_BATCH 32
#define C_DIM   256
#define C8_DIM  32
#define N_DIM   1024
#define O_DIM   320   // 32 (q) + 32 (k) + 256 (v) fused projection rows

// ---- workspace layout (bytes), all 16B+ aligned ----
#define OFF_XBT 0ull                                   // bf16 [32][1024][256]
#define OFF_WB  (OFF_XBT + 32ull*1024*256*2)           // bf16 [320][256]
#define OFF_BC  (OFF_WB  + 320ull*256*2)               // f32  [320] (+pad)
#define OFF_QT  (OFF_BC  + 512ull*4)                   // bf16 [32][1024][32]
#define OFF_KT  (OFF_QT  + 32ull*1024*32*2)            // bf16 [32][1024][32]
#define OFF_VB  (OFF_KT  + 32ull*1024*32*2)            // bf16 [32][256][1024]
#define OFF_RM  (OFF_VB  + 32ull*256*1024*2)           // f32  [32][1024] row max
#define OFF_RZ  (OFF_RM  + 32ull*1024*4)               // f32  [32][1024] 1/rowsum

__device__ __forceinline__ v16bf cat16(v8bf lo, v8bf hi) {
  return __builtin_shufflevector(lo, hi, 0,1,2,3,4,5,6,7,8,9,10,11,12,13,14,15);
}
// A-fragment (16-bit 16x32): lane holds row M=lane&15; 16 contiguous K at half*16
__device__ __forceinline__ v16bf load_afrag(const __bf16* rowbase_plus_koff) {
  v8bf lo = *(const v8bf*)(rowbase_plus_koff);
  v8bf hi = *(const v8bf*)(rowbase_plus_koff + 8);
  return cat16(lo, hi);
}
// B-fragment (16-bit 32x16): lane holds col N=lane&15; 16 contiguous K at half*16
__device__ __forceinline__ v16bf load_bfrag(const __bf16* colbase_plus_koff) {
  v8bf lo = *(const v8bf*)(colbase_plus_koff);
  v8bf hi = *(const v8bf*)(colbase_plus_koff + 8);
  return cat16(lo, hi);
}
__device__ __forceinline__ v8f wmma_bf16(v16bf a, v16bf b, v8f c) {
  return __builtin_amdgcn_wmma_f32_16x16x32_bf16(false, a, false, b, (short)0, c, false, false);
}

// ---------- kernel 0a: x (f32 [32][256][1024]) -> XbT (bf16 [32][1024][256]) ----------
__global__ void k_cvt_x(const float* __restrict__ x, __bf16* __restrict__ xbt) {
  unsigned i = blockIdx.x * 256u + threadIdx.x;           // 8388608 total, exact
  unsigned n = i & 1023u;
  unsigned mc = i >> 10;
  unsigned c = mc & 255u;
  unsigned m = mc >> 8;
  xbt[((unsigned long long)(m * 1024u + n)) * 256u + c] = (__bf16)x[i];
}

// ---------- kernel 0b: pack [wq;wk;wv] -> Wb bf16, biases -> bcat f32 ----------
__global__ void k_cvt_w(const float* __restrict__ wq, const float* __restrict__ wk,
                        const float* __restrict__ wv, const float* __restrict__ bq,
                        const float* __restrict__ bk, const float* __restrict__ bv,
                        __bf16* __restrict__ wb, float* __restrict__ bc) {
  int i = blockIdx.x * 256 + threadIdx.x;
  if (i < O_DIM * C_DIM) {
    int o = i >> 8, c = i & 255;
    float v;
    if (o < 32)       v = wq[o * 256 + c];
    else if (o < 64)  v = wk[(o - 32) * 256 + c];
    else              v = wv[(o - 64) * 256 + c];
    wb[i] = (__bf16)v;
  }
  if (i < O_DIM) {
    bc[i] = (i < 32) ? bq[i] : ((i < 64) ? bk[i - 32] : bv[i - 64]);
  }
}

// ---------- kernel 1: fused projection GEMM  [320x256] x [256x1024] per m ----------
__global__ void __launch_bounds__(256) k_proj(const __bf16* __restrict__ xbt,
                                              const __bf16* __restrict__ wb,
                                              const float* __restrict__ bc,
                                              __bf16* __restrict__ qt,
                                              __bf16* __restrict__ kt,
                                              __bf16* __restrict__ vb) {
  const int lane = threadIdx.x & 31;
  const int wid  = blockIdx.x * 8 + (threadIdx.x >> 5);   // 40960 waves
  const int nt   = wid & 63;
  const int rest = wid >> 6;
  const int ot   = rest % 20;
  const int m    = rest / 20;
  const int half = lane >> 4, l15 = lane & 15;
  const int obase = ot * 16, nbase = nt * 16;
  const int ncol = nbase + l15;

  const __bf16* arow = wb + (obase + l15) * C_DIM + half * 16;
  const __bf16* brow = xbt + ((unsigned long long)m * N_DIM + ncol) * C_DIM + half * 16;

  v8f acc = {0.f, 0.f, 0.f, 0.f, 0.f, 0.f, 0.f, 0.f};
#pragma unroll
  for (int cb = 0; cb < C_DIM; cb += 32) {
    v16bf a = load_afrag(arow + cb);
    v16bf b = load_bfrag(brow + cb);
    acc = wmma_bf16(a, b, acc);
  }
#pragma unroll
  for (int g = 0; g < 8; ++g) {
    int o = obase + half * 8 + g;
    float val = acc[g] + bc[o];
    __bf16 h = (__bf16)val;
    if (obase < 32)
      qt[((unsigned long long)m * N_DIM + ncol) * C8_DIM + o] = h;
    else if (obase < 64)
      kt[((unsigned long long)m * N_DIM + ncol) * C8_DIM + (o - 32)] = h;
    else
      vb[((unsigned long long)m * C_DIM + (o - 64)) * N_DIM + ncol] = h;
  }
}

// ---------- kernel 2: per-row softmax stats  M[n], 1/Z[n]  (E = Qt x Kt^T) ----------
__global__ void __launch_bounds__(256) k_stats(const __bf16* __restrict__ qt,
                                               const __bf16* __restrict__ kt,
                                               float* __restrict__ rowm,
                                               float* __restrict__ rowrz) {
  const int lane = threadIdx.x & 31;
  const int wid  = blockIdx.x * 8 + (threadIdx.x >> 5);   // 2048 waves
  const int nt = wid & 63;
  const int m  = wid >> 6;
  const int half = lane >> 4, l15 = lane & 15;

  const __bf16* qtm = qt + (unsigned long long)m * N_DIM * C8_DIM;
  const __bf16* ktm = kt + (unsigned long long)m * N_DIM * C8_DIM;
  v16bf a = load_afrag(qtm + (nt * 16 + l15) * C8_DIM + half * 16);   // invariant A

  float rm[8], rz[8];
#pragma unroll
  for (int g = 0; g < 8; ++g) { rm[g] = -1.0e30f; rz[g] = 0.f; }

  for (int j = 0; j < N_DIM; j += 16) {
    v16bf b = load_bfrag(ktm + (j + l15) * C8_DIM + half * 16);
    v8f z = {0.f, 0.f, 0.f, 0.f, 0.f, 0.f, 0.f, 0.f};
    v8f e = wmma_bf16(a, b, z);
#pragma unroll
    for (int g = 0; g < 8; ++g) {
      float ev = e[g];
      float nm = fmaxf(rm[g], ev);
      rz[g] = rz[g] * __expf(rm[g] - nm) + __expf(ev - nm);
      rm[g] = nm;
    }
  }
  // combine across the 16 lanes sharing each row (xor <= 8 stays in-half)
#pragma unroll
  for (int off = 1; off <= 8; off <<= 1) {
#pragma unroll
    for (int g = 0; g < 8; ++g) {
      float om = __shfl_xor(rm[g], off, 32);
      float oz = __shfl_xor(rz[g], off, 32);
      float nm = fmaxf(rm[g], om);
      rz[g] = rz[g] * __expf(rm[g] - nm) + oz * __expf(om - nm);
      rm[g] = nm;
    }
  }
  if (l15 == 0) {
#pragma unroll
    for (int g = 0; g < 8; ++g) {
      int n = nt * 16 + half * 8 + g;
      rowm[m * N_DIM + n]  = rm[g];
      rowrz[m * N_DIM + n] = 1.0f / rz[g];
    }
  }
}

// ---------- kernel 3: out = gamma * (V @ softmax(E)) + x ----------
__global__ void __launch_bounds__(256) k_attn_out(const __bf16* __restrict__ qt,
                                                  const __bf16* __restrict__ kt,
                                                  const __bf16* __restrict__ vb,
                                                  const float* __restrict__ rowm,
                                                  const float* __restrict__ rowrz,
                                                  const float* __restrict__ x,
                                                  const float* __restrict__ gamma,
                                                  float* __restrict__ out) {
  const int lane = threadIdx.x & 31;
  const int w    = threadIdx.x >> 5;     // 8 waves: 4 j-tiles x 2 c-halves
  const int jb   = blockIdx.x & 15;      // 16 j-blocks of 64 columns
  const int m    = blockIdx.x >> 4;
  const int half = lane >> 4, l15 = lane & 15;
  const int jtile  = jb * 64 + (w >> 1) * 16;
  const int cbase0 = (w & 1) * 128;
  const int jcol   = jtile + l15;

  const __bf16* qtm = qt + (unsigned long long)m * N_DIM * C8_DIM;
  const __bf16* ktm = kt + (unsigned long long)m * N_DIM * C8_DIM;
  const __bf16* vbm = vb + (unsigned long long)m * C_DIM * N_DIM;
  const float*  rmm = rowm + m * N_DIM;
  const float*  rzm = rowrz + m * N_DIM;

  v16bf bk = load_bfrag(ktm + jcol * C8_DIM + half * 16);   // invariant B for E

  v8f acc[8];
#pragma unroll
  for (int cc = 0; cc < 8; ++cc) acc[cc] = (v8f){0.f,0.f,0.f,0.f,0.f,0.f,0.f,0.f};

  for (int n = 0; n < N_DIM; n += 32) {
    v16bf aql = load_afrag(qtm + (n + l15) * C8_DIM + half * 16);
    v16bf aqh = load_afrag(qtm + (n + 16 + l15) * C8_DIM + half * 16);
    v8f z = {0.f, 0.f, 0.f, 0.f, 0.f, 0.f, 0.f, 0.f};
    v8f elo = wmma_bf16(aql, bk, z);
    v8f ehi = wmma_bf16(aqh, bk, z);

    float plo[8], phi[8];
    const int nb = n + half * 8;
#pragma unroll
    for (int g = 0; g < 8; ++g) {
      int r = nb + g;
      plo[g] = __expf(elo[g] - rmm[r]) * rzm[r];
      int r2 = r + 16;
      phi[g] = __expf(ehi[g] - rmm[r2]) * rzm[r2];
    }
    // D-layout -> B-fragment of P: lane-half swap per register pair
    v16bf bp;
#pragma unroll
    for (int g = 0; g < 4; ++g) {
      float sh0 = __shfl_xor(phi[2 * g], 16, 32);
      float sh1 = __shfl_xor(phi[2 * g + 1], 16, 32);
      float v0 = (lane < 16) ? plo[2 * g]     : sh0;
      float v1 = (lane < 16) ? plo[2 * g + 1] : sh1;
      bp[2 * g]     = (__bf16)v0;
      bp[2 * g + 1] = (__bf16)v1;
    }
#pragma unroll
    for (int g = 4; g < 8; ++g) {
      int idx = 2 * g - 8;
      float sh0 = __shfl_xor(plo[idx], 16, 32);
      float sh1 = __shfl_xor(plo[idx + 1], 16, 32);
      float v0 = (lane < 16) ? sh0 : phi[idx];
      float v1 = (lane < 16) ? sh1 : phi[idx + 1];
      bp[2 * g]     = (__bf16)v0;
      bp[2 * g + 1] = (__bf16)v1;
    }
    // accumulate out_tile += V_frag(16c x 32n) @ P_frag(32n x 16j)
#pragma unroll
    for (int cc = 0; cc < 8; ++cc) {
      int crow = cbase0 + cc * 16 + l15;
      const __bf16* vrow = vbm + (unsigned long long)crow * N_DIM + n + half * 16;
      v16bf av = load_afrag(vrow);
      acc[cc] = wmma_bf16(av, bp, acc[cc]);
    }
  }

  const float g0 = gamma[0];
#pragma unroll
  for (int cc = 0; cc < 8; ++cc) {
#pragma unroll
    for (int g = 0; g < 8; ++g) {
      int c = cbase0 + cc * 16 + half * 8 + g;
      unsigned long long idx = ((unsigned long long)m * C_DIM + c) * N_DIM + jcol;
      out[idx] = g0 * acc[cc][g] + x[idx];
    }
  }
}

extern "C" void kernel_launch(void* const* d_in, const int* in_sizes, int n_in,
                              void* d_out, int out_size, void* d_ws, size_t ws_size,
                              hipStream_t stream) {
  const float* x     = (const float*)d_in[0];
  const float* wq    = (const float*)d_in[1];
  const float* bq    = (const float*)d_in[2];
  const float* wk    = (const float*)d_in[3];
  const float* bk    = (const float*)d_in[4];
  const float* wv    = (const float*)d_in[5];
  const float* bv    = (const float*)d_in[6];
  const float* gamma = (const float*)d_in[7];
  float* out = (float*)d_out;

  char* ws = (char*)d_ws;
  __bf16* xbt = (__bf16*)(ws + OFF_XBT);
  __bf16* wb  = (__bf16*)(ws + OFF_WB);
  float*  bc  = (float*)(ws + OFF_BC);
  __bf16* qt  = (__bf16*)(ws + OFF_QT);
  __bf16* kt  = (__bf16*)(ws + OFF_KT);
  __bf16* vb  = (__bf16*)(ws + OFF_VB);
  float*  rm  = (float*)(ws + OFF_RM);
  float*  rz  = (float*)(ws + OFF_RZ);

  k_cvt_x<<<32768, 256, 0, stream>>>(x, xbt);
  k_cvt_w<<<320, 256, 0, stream>>>(wq, wk, wv, bq, bk, bv, wb, bc);
  k_proj<<<5120, 256, 0, stream>>>(xbt, wb, bc, qt, kt, vb);
  k_stats<<<256, 256, 0, stream>>>(qt, kt, rm, rz);
  k_attn_out<<<512, 256, 0, stream>>>(qt, kt, vb, rm, rz, x, gamma, out);
}
